// MaxxVitDecoderBlock_84585085927949
// MI455X (gfx1250) — matrix operand order repaired and checked by reading, not compiled, not run on hardware
//
// MaxViT decoder block for MI455X (gfx1250, wave32, WMMA + TDM).
//
// Precision strategy: all heavy GEMMs run v_wmma_f32_16x16x32_bf16 (A converted
// f32->bf16 during LDS staging, weights pre-converted once, f32 accumulate).
// Weight tiles are DMA'd into LDS by the Tensor Data Mover (tensor_load_to_lds,
// double-buffered, padded rows 64B data + 16B pad == our 40-bf16 LDS stride).
// Elementwise stages (BN/GELU/dw-conv/SE/LN/softmax/layout) stay fp32 and are
// bandwidth-bound at 23.3 TB/s.
//
// Assumed d_in order = JAX pytree flatten (dict keys sorted, recursive):
//  0 b3  1 b_se1  2 b_se2
//  3..15  blk: b_fc1 b_fc2 b_proj b_qkv norm1.b norm1.g norm2.b norm2.g
//              rel_bias w_fc1 w_fc2 w_proj w_qkv
// 16..19  bn1: b g m v      20..23 bn2: b g m v
// 24..36  grd: (same layout as blk)
// 37..40  pre_bn: b g m v   41 up_b  42 up_w  43 w1  44 w3  45 w_dw
// 46 w_se1  47 w_se2  48 rel_idx (int32)  49 x (f32 NCHW 16x256x28x28)

#include <hip/hip_runtime.h>
#include <cstddef>
#include <cstdint>

#define USE_TDM 1   // tensor_load_to_lds for GEMM weight tiles (revert switch)

typedef __bf16 bf16_t;
typedef __attribute__((ext_vector_type(16))) __bf16 v16bf;
typedef __attribute__((ext_vector_type(8)))  __bf16 v8bf;
typedef __attribute__((ext_vector_type(4)))  __bf16 v4bf;
typedef __attribute__((ext_vector_type(8)))  float  v8f;
typedef __attribute__((ext_vector_type(4)))  int    i32x4;
typedef __attribute__((ext_vector_type(8)))  int    i32x8;

#define B_     16
#define H_     56
#define W_     56
#define C_     128
#define CIN_   256
#define MID_   512
#define HEADS_ 4
#define DH_    32
#define HW_    (H_ * W_)          // 3136
#define M1_    (B_ * HW_)         // 50176 tokens at 56x56
#define M0_    (B_ * 28 * 28)     // 12544 tokens at 28x28
#define NWIN_  (B_ * 64)          // 1024 windows (8x8 of 7x7, both partitions)
#define EPS_   1e-5f
#define LDT_   40                 // LDS row stride (bf16): 64B data + 16B pad

__device__ __forceinline__ float gelu_f(float x) {
  return 0.5f * x * (1.0f + erff(x * 0.70710678118654752440f));
}

#if USE_TDM
// TDM 2D bf16 tile load: rows x cols elements, global row stride strideElems,
// written to LDS contiguously with +16B padding after every 64B of data
// (pad_interval=3 -> 16 DWORDs, pad_amount=3 -> 4 DWORDs), matching LDT_=40.
// D# per CDNA5 ISA ch.8: group0 = {count, lds_addr, global_addr, type=2},
// group1 = {flags/pad, tensor dims, tile dims, dim0 stride}. 2D tensor ->
// groups 2/3 NULL -> two-SGPR-group form of the instruction.
__device__ __forceinline__ void tdm_load_bf16_tile(const bf16_t* g,
                                                   unsigned ldsByteOff, int rows,
                                                   int cols, int strideElems) {
  const unsigned long long ga = (unsigned long long)(uintptr_t)g;
  i32x4 g0;
  g0[0] = 1;                                            // count=1, user desc
  g0[1] = (int)ldsByteOff;                              // lds_addr
  g0[2] = (int)(unsigned)(ga & 0xFFFFFFFFull);          // global_addr[31:0]
  g0[3] = (int)(((ga >> 32) & 0x1FFFFFFull) | (2u << 30));  // addr[56:32]|type=2
  i32x8 g1;
  g1[0] = (int)((1u << 16) | (1u << 20) | (3u << 22) | (3u << 25));
  //            data_size=2B  pad_enable  pad_interval=64B  pad_amount=16B
  g1[1] = (int)(((unsigned)cols & 0xFFFFu) << 16);      // tensor_dim0[15:0]
  g1[2] = (int)(((unsigned)cols >> 16) | (((unsigned)rows & 0xFFFFu) << 16));
  g1[3] = (int)(((unsigned)rows >> 16) | ((unsigned)cols << 16));  // tile_dim0
  g1[4] = rows & 0xFFFF;                                // tile_dim1 (tile_dim2=0)
  g1[5] = strideElems;                                  // tensor_dim0_stride lo32
  g1[6] = 0;
  g1[7] = 0;
  asm volatile("tensor_load_to_lds %0, %1" ::"s"(g0), "s"(g1) : "memory");
}
#endif

// A fragment per ISA 7.12.2 (16-bit A 16x32): lanes 0-15 K{0..7,16..23},
// lanes 16-31 K{8..15,24..31}.  row = &LDS[m * stride] (+ kstep base).
__device__ __forceinline__ v16bf make_afrag(const bf16_t* row, int k0a) {
  v8bf lo = *(const v8bf*)(row + k0a);
  v8bf hi = *(const v8bf*)(row + k0a + 16);
  v16bf f;
#pragma unroll
  for (int e = 0; e < 8; ++e) { f[e] = lo[e]; f[e + 8] = hi[e]; }
  return f;
}
// B fragment (16-bit B 32x16): lane n holds B[k][n]=W[n][k], lanes 0-15 K{0..7,16..23},
// lanes 16-31 K{8..15,24..31} (same order as A fragment).
__device__ __forceinline__ v16bf make_bfrag(const bf16_t* row, int k0b) {
  v8bf lo = *(const v8bf*)(row + k0b);
  v8bf hi = *(const v8bf*)(row + k0b + 16);
  v16bf f;
#pragma unroll
  for (int e = 0; e < 8; ++e) { f[e] = lo[e]; f[e + 8] = hi[e]; }
  return f;
}

// ---------------------------------------------------------------------------
// Generic WMMA GEMM: Out[M,N] = epi( A[M,K](f32) * Wt[N,K](bf16)^T )
// epilogue: +bias[n]; *scale[n]+shift[n]; gelu; += addsrc[m,n]
// ASE (template): per-(row-group, k) A scaling (SE), group = m / aseDiv.
// M % 128 == 0, N % 128 == 0, K % 32 == 0 (holds for every call site).
// ---------------------------------------------------------------------------
template <bool ASE>
__global__ __launch_bounds__(256) void gemm_bf16(
    const float* __restrict__ A, const bf16_t* __restrict__ Wt,
    float* __restrict__ Out,
    const float* __restrict__ bias, const float* __restrict__ scale,
    const float* __restrict__ shift, const float* __restrict__ addsrc,
    const float* __restrict__ ase, int M, int N, int K, int aseDiv, int doGelu) {
  __shared__ __align__(16) bf16_t As[128 * LDT_];
#if USE_TDM
  __shared__ __align__(16) bf16_t Bs[2][128 * LDT_];
#else
  __shared__ __align__(16) bf16_t Bs[1][128 * LDT_];
#endif
  const int tid  = threadIdx.x;
  const int lane = tid & 31;
  const int wave = tid >> 5;
  const int wm   = wave >> 1;   // 0..3 : 32-row strip
  const int wn   = wave & 1;    // 0..1 : 64-col strip
  const int m0   = blockIdx.x * 128;
  const int n0   = blockIdx.y * 128;
  const int k0a  = (lane >= 16) ? 8 : 0;
  const int k0b  = (lane >= 16) ? 8 : 0;

  v8f acc[2][4];
#pragma unroll
  for (int i = 0; i < 2; ++i)
#pragma unroll
    for (int j = 0; j < 4; ++j)
#pragma unroll
      for (int r = 0; r < 8; ++r) acc[i][j][r] = 0.0f;

  const int ksteps = K / 32;
#if USE_TDM
  if (wave == 0)   // DMA first weight tile (wave-level op; EXEC ignored)
    tdm_load_bf16_tile(Wt + (size_t)n0 * K, (unsigned)(uintptr_t)&Bs[0][0],
                       128, 32, K);
#endif
  for (int kt = 0; kt < ksteps; ++kt) {
    const int k0  = kt * 32;
    const int cur = USE_TDM ? (kt & 1) : 0;
    // stage A (128x32 f32 -> bf16) via registers
    float4 areg[4], sreg[4];
#pragma unroll
    for (int i = 0; i < 4; ++i) {
      const int lin = tid + i * 256;       // float4 units
      const int row = lin >> 3, c4 = lin & 7;
      areg[i] = *(const float4*)(A + (size_t)(m0 + row) * K + k0 + c4 * 4);
      if (ASE)
        sreg[i] = *(const float4*)(ase + (size_t)((m0 + row) / aseDiv) * K + k0 + c4 * 4);
    }
#if !USE_TDM
    uint4 breg[2];
#pragma unroll
    for (int i = 0; i < 2; ++i) {
      const int lin = tid + i * 256;       // 8-bf16 chunk units
      const int row = lin >> 2, c8 = lin & 3;
      breg[i] = *(const uint4*)(Wt + (size_t)(n0 + row) * K + k0 + c8 * 8);
    }
#endif
    __syncthreads();                        // previous tile fully consumed
#if USE_TDM
    if (wave == 0 && kt + 1 < ksteps)       // overlap: DMA next tile into alt buf
      tdm_load_bf16_tile(Wt + (size_t)n0 * K + (kt + 1) * 32,
                         (unsigned)(uintptr_t)&Bs[(kt + 1) & 1][0], 128, 32, K);
#endif
#pragma unroll
    for (int i = 0; i < 4; ++i) {
      const int lin = tid + i * 256;
      const int row = lin >> 3, c4 = lin & 7;
      float4 a = areg[i];
      if (ASE) { a.x *= sreg[i].x; a.y *= sreg[i].y; a.z *= sreg[i].z; a.w *= sreg[i].w; }
      v4bf cv;
      cv[0] = (bf16_t)a.x; cv[1] = (bf16_t)a.y; cv[2] = (bf16_t)a.z; cv[3] = (bf16_t)a.w;
      *(v4bf*)&As[row * LDT_ + c4 * 4] = cv;
    }
#if USE_TDM
    if (wave == 0) {                        // TDM completes in order: cnt<=1 ->
      if (kt + 1 < ksteps)                  // current tile is resident
        __builtin_amdgcn_s_wait_tensorcnt(1);
      else
        __builtin_amdgcn_s_wait_tensorcnt(0);
    }
#else
#pragma unroll
    for (int i = 0; i < 2; ++i) {
      const int lin = tid + i * 256;
      const int row = lin >> 2, c8 = lin & 3;
      *(uint4*)&Bs[0][row * LDT_ + c8 * 8] = breg[i];
    }
#endif
    __syncthreads();
    if (kt + 1 < ksteps) {                  // CDNA5 global_prefetch_b8 on A stream
      const int row = tid >> 3, c4 = tid & 7;
      __builtin_prefetch(A + (size_t)(m0 + row) * K + k0 + 32 + c4 * 4, 0, 0);
    }
    v16bf afrag[2];
#pragma unroll
    for (int mt = 0; mt < 2; ++mt)
      afrag[mt] = make_afrag(&As[(wm * 32 + mt * 16 + (lane & 15)) * LDT_], k0a);
#pragma unroll
    for (int nt = 0; nt < 4; ++nt) {
      v16bf bfrag = make_bfrag(&Bs[cur][(wn * 64 + nt * 16 + (lane & 15)) * LDT_], k0b);
#pragma unroll
      for (int mt = 0; mt < 2; ++mt)
        acc[mt][nt] = __builtin_amdgcn_wmma_f32_16x16x32_bf16(
            false, afrag[mt], false, bfrag, (short)0, acc[mt][nt], false, false);
    }
  }
  // epilogue: C/D layout -> VGPR r: lanes 0-15 M=r, lanes 16-31 M=r+8; N=lane&15
  const int nb = n0 + wn * 64;
  const int mb = m0 + wm * 32 + ((lane >= 16) ? 8 : 0);
#pragma unroll
  for (int nt = 0; nt < 4; ++nt) {
    const int n   = nb + nt * 16 + (lane & 15);
    const float bia = bias ? bias[n] : 0.0f;
    const float sc  = scale ? scale[n] : 1.0f;
    const float sh  = shift ? shift[n] : 0.0f;
#pragma unroll
    for (int mt = 0; mt < 2; ++mt)
#pragma unroll
      for (int r = 0; r < 8; ++r) {
        const int m = mb + mt * 16 + r;
        float v = acc[mt][nt][r] + bia;
        if (scale) v = v * sc + sh;
        if (doGelu) v = gelu_f(v);
        if (addsrc) v += addsrc[(size_t)m * N + n];
        Out[(size_t)m * N + n] = v;
      }
  }
}

// ---------------------------------------------------------------------------
// Windowed MHSA: one WG (4 waves) per (window, head). 49 tokens padded to 64.
// qkv: [M1,384] f32 (ch = which*128 + head*32 + d).  out: [M1,128] f32.
// ---------------------------------------------------------------------------
__global__ __launch_bounds__(128) void attn_kernel(
    const float* __restrict__ qkv, const float* __restrict__ biasTab,
    float* __restrict__ out, int gridPart) {
  __shared__ __align__(16) bf16_t Qs[64 * LDT_];
  __shared__ __align__(16) bf16_t Ks[64 * LDT_];
  __shared__ __align__(16) bf16_t Vt[32 * 72];   // V transposed: [d][token]
  __shared__ __align__(16) float  Sf[64 * 68];
  __shared__ __align__(16) bf16_t Ps[64 * 72];

  const int u  = blockIdx.x;        // window 0..1023
  const int hd = blockIdx.y;        // head 0..3
  const int b  = u >> 6;
  const int wh = (u >> 3) & 7;
  const int ww = u & 7;
  const int tid = threadIdx.x, lane = tid & 31, wave = tid >> 5;

  auto tok = [&](int e) -> int {    // token index of window element e (0..48)
    const int i = e / 7, j = e % 7;
    const int hh = gridPart ? (i * 8 + wh) : (wh * 7 + i);
    const int xx = gridPart ? (j * 8 + ww) : (ww * 7 + j);
    return b * HW_ + hh * W_ + xx;
  };

  // load q (pre-scaled by 1/sqrt(DH)), k, v ; zero-pad tokens 49..63
  for (int idx = tid; idx < 64 * 32; idx += 128) {
    const int e = idx >> 5, d = idx & 31;
    float qv = 0.f, kv = 0.f, vv = 0.f;
    if (e < 49) {
      const size_t base = (size_t)tok(e) * 384 + hd * 32 + d;
      qv = qkv[base] * 0.17677669529663687f;
      kv = qkv[base + 128];
      vv = qkv[base + 256];
    }
    Qs[e * LDT_ + d] = (bf16_t)qv;
    Ks[e * LDT_ + d] = (bf16_t)kv;
    Vt[d * 72 + e]   = (bf16_t)vv;
  }
  __syncthreads();

  // S = Q K^T  (64x64, K=32, one WMMA step per 16x16 tile)
  {
    const int k0a = (lane >= 16) ? 8 : 0;
    const int k0b = (lane >= 16) ? 8 : 0;
    const v16bf af = make_afrag(&Qs[(wave * 16 + (lane & 15)) * LDT_], k0a);
#pragma unroll
    for (int nt = 0; nt < 4; ++nt) {
      v16bf bf = make_bfrag(&Ks[(nt * 16 + (lane & 15)) * LDT_], k0b);
      v8f s;
#pragma unroll
      for (int r = 0; r < 8; ++r) s[r] = 0.0f;
      s = __builtin_amdgcn_wmma_f32_16x16x32_bf16(false, af, false, bf, (short)0,
                                                  s, false, false);
#pragma unroll
      for (int r = 0; r < 8; ++r) {
        const int q = wave * 16 + r + ((lane >= 16) ? 8 : 0);
        Sf[q * 68 + nt * 16 + (lane & 15)] = s[r];
      }
    }
  }
  __syncthreads();

  // softmax rows (rel-pos bias added), P in bf16; pad cols/rows with zeros
  const float* bt = biasTab + (size_t)hd * 49 * 49;
  for (int q = tid; q < 64; q += 128) {
    if (q < 49) {
      float mx = -1e30f;
      for (int k2 = 0; k2 < 49; ++k2) {
        const float s = Sf[q * 68 + k2] + bt[q * 49 + k2];
        Sf[q * 68 + k2] = s;
        mx = fmaxf(mx, s);
      }
      float sum = 0.f;
      for (int k2 = 0; k2 < 49; ++k2) {
        const float e2 = __expf(Sf[q * 68 + k2] - mx);
        Sf[q * 68 + k2] = e2;
        sum += e2;
      }
      const float inv = 1.0f / sum;
      for (int k2 = 0; k2 < 49; ++k2) Ps[q * 72 + k2] = (bf16_t)(Sf[q * 68 + k2] * inv);
      for (int k2 = 49; k2 < 64; ++k2) Ps[q * 72 + k2] = (bf16_t)0.0f;
    } else {
      for (int k2 = 0; k2 < 64; ++k2) Ps[q * 72 + k2] = (bf16_t)0.0f;
    }
  }
  __syncthreads();

  // O = P V  (64x32, K=64 -> two WMMA K-steps)
  {
    const int k0a = (lane >= 16) ? 8 : 0;
    const int m   = wave * 16 + (lane & 15);
    v8f o[2];
#pragma unroll
    for (int nt = 0; nt < 2; ++nt)
#pragma unroll
      for (int r = 0; r < 8; ++r) o[nt][r] = 0.0f;
#pragma unroll
    for (int ks = 0; ks < 64; ks += 32) {
      const v16bf af = make_afrag(&Ps[m * 72 + ks], k0a);
#pragma unroll
      for (int nt = 0; nt < 2; ++nt) {
        const int k0b = ks + ((lane >= 16) ? 8 : 0);
        v16bf bf = make_bfrag(&Vt[(nt * 16 + (lane & 15)) * 72], k0b);
        o[nt] = __builtin_amdgcn_wmma_f32_16x16x32_bf16(false, af, false, bf,
                                                        (short)0, o[nt], false, false);
      }
    }
#pragma unroll
    for (int nt = 0; nt < 2; ++nt)
#pragma unroll
      for (int r = 0; r < 8; ++r) {
        const int q = wave * 16 + r + ((lane >= 16) ? 8 : 0);
        if (q < 49)
          out[(size_t)tok(q) * C_ + hd * 32 + nt * 16 + (lane & 15)] = o[nt][r];
      }
  }
}

// ---------------------------------------------------------------------------
// Elementwise / small kernels
// ---------------------------------------------------------------------------
__global__ __launch_bounds__(256) void ln_kernel(
    const float* __restrict__ x, float* __restrict__ y,
    const float* __restrict__ g, const float* __restrict__ bb, int M) {
  const int wave = threadIdx.x >> 5, lane = threadIdx.x & 31;
  const int t = blockIdx.x * 8 + wave;
  if (t >= M) return;
  const float4 v = *(const float4*)(x + (size_t)t * C_ + lane * 4);
  float s  = v.x + v.y + v.z + v.w;
  float s2 = v.x * v.x + v.y * v.y + v.z * v.z + v.w * v.w;
#pragma unroll
  for (int m = 16; m >= 1; m >>= 1) {
    s  += __shfl_xor(s, m, 32);
    s2 += __shfl_xor(s2, m, 32);
  }
  const float mu  = s * (1.0f / 128.0f);
  const float var = s2 * (1.0f / 128.0f) - mu * mu;
  const float inv = rsqrtf(var + EPS_);
  const float4 gg = *(const float4*)(g + lane * 4);
  const float4 bv = *(const float4*)(bb + lane * 4);
  float4 o;
  o.x = (v.x - mu) * inv * gg.x + bv.x;
  o.y = (v.y - mu) * inv * gg.y + bv.y;
  o.z = (v.z - mu) * inv * gg.z + bv.z;
  o.w = (v.w - mu) * inv * gg.w + bv.w;
  *(float4*)(y + (size_t)t * C_ + lane * 4) = o;
}

__global__ __launch_bounds__(256) void dwconv_kernel(
    const float* __restrict__ in, const float* __restrict__ wdw,
    const float* __restrict__ sc, const float* __restrict__ sh,
    float* __restrict__ outp) {
  const int t = blockIdx.x;
  const int b = t / HW_, p = t % HW_, hh = p / W_, ww = p % W_;
  for (int c = threadIdx.x; c < MID_; c += 256) {
    float acc = 0.f;
#pragma unroll
    for (int dy = -1; dy <= 1; ++dy)
#pragma unroll
      for (int dx = -1; dx <= 1; ++dx) {
        const int y2 = hh + dy, x2 = ww + dx;
        if ((unsigned)y2 < (unsigned)H_ && (unsigned)x2 < (unsigned)W_)
          acc += in[((size_t)b * HW_ + y2 * W_ + x2) * MID_ + c] *
                 wdw[c * 9 + (dy + 1) * 3 + (dx + 1)];
      }
    outp[(size_t)t * MID_ + c] = gelu_f(acc * sc[c] + sh[c]);
  }
}

__global__ __launch_bounds__(128) void pool_kernel(const float* __restrict__ in,
                                                   float* __restrict__ pooled) {
  const int b = blockIdx.x;
  const int c = blockIdx.y * 128 + threadIdx.x;
  float s = 0.f;
  for (int p = 0; p < HW_; ++p) s += in[((size_t)b * HW_ + p) * MID_ + c];
  pooled[b * MID_ + c] = s * (1.0f / (float)HW_);
}

__global__ __launch_bounds__(512) void se_kernel(
    const float* __restrict__ pooled, const float* __restrict__ w1,
    const float* __restrict__ b1, const float* __restrict__ w2,
    const float* __restrict__ b2, float* __restrict__ se) {
  __shared__ float s1[32];
  const int b = blockIdx.x, tid = threadIdx.x;
  if (tid < 32) {
    float a = b1[tid];
    for (int k = 0; k < MID_; ++k) a += pooled[b * MID_ + k] * w1[tid * MID_ + k];
    s1[tid] = a / (1.0f + __expf(-a));                  // silu
  }
  __syncthreads();
  float a = b2[tid];
  for (int r = 0; r < 32; ++r) a += s1[r] * w2[tid * 32 + r];
  se[b * MID_ + tid] = 1.0f / (1.0f + __expf(-a));      // sigmoid
}

__global__ void cvt_bf16_kernel(const float* __restrict__ w, bf16_t* __restrict__ o,
                                const float* __restrict__ kscale, int total, int K) {
  const int idx = blockIdx.x * 256 + threadIdx.x;
  if (idx >= total) return;
  float v = w[idx];
  if (kscale) v *= kscale[idx % K];
  o[idx] = (bf16_t)v;
}

// up_w [c=256][d=128][i=2][j=2] -> Wup[n=(i*2+j)*128+d][c]  (bf16)
__global__ void upw_pack_kernel(const float* __restrict__ upw, bf16_t* __restrict__ o) {
  const int idx = blockIdx.x * 256 + threadIdx.x;
  if (idx >= 512 * 256) return;
  const int n = idx >> 8, c = idx & 255;
  const int ij = n >> 7, d = n & 127, i = ij >> 1, j = ij & 1;
  o[idx] = (bf16_t)upw[((c * 128 + d) * 2 + i) * 2 + j];
}

__global__ void bnvec_kernel(const float* g, const float* bb, const float* m,
                             const float* v, float* sc, float* sh, int Cn) {
  const int c = blockIdx.x * 256 + threadIdx.x;
  if (c >= Cn) return;
  const float s = g[c] * rsqrtf(v[c] + EPS_);
  sc[c] = s;
  sh[c] = bb[c] - m[c] * s;
}

// pre_bn shift folded through w1: b1eff[n] = sum_k (pre_b - pre_m*pre_s)[k] * w1[n][k]
__global__ void b1eff_kernel(const float* __restrict__ w1, const float* pg,
                             const float* pb, const float* pm, const float* pv,
                             float* __restrict__ b1eff) {
  const int n = blockIdx.x * 256 + threadIdx.x;
  if (n >= MID_) return;
  float a = 0.f;
  for (int k = 0; k < C_; ++k) {
    const float s = pg[k] * rsqrtf(pv[k] + EPS_);
    a += (pb[k] - pm[k] * s) * w1[n * C_ + k];
  }
  b1eff[n] = a;
}

__global__ void biastab_kernel(const float* __restrict__ rb,
                               const int* __restrict__ ridx, float* __restrict__ o) {
  const int idx = blockIdx.x * 256 + threadIdx.x;      // h*2401 + q*49 + k
  if (idx >= HEADS_ * 2401) return;
  const int h = idx / 2401, qk = idx % 2401;
  o[idx] = rb[ridx[qk] * HEADS_ + h];
}

// NCHW 16x256x28x28 -> token-major [M0,256]
__global__ void nchw2nhwc_kernel(const float* __restrict__ x, float* __restrict__ xt) {
  const int idx = blockIdx.x * 256 + threadIdx.x;
  if (idx >= M0_ * CIN_) return;
  const int t = idx >> 8, c = idx & 255;
  const int b = t / 784, p = t % 784;
  xt[idx] = x[(size_t)b * (CIN_ * 784) + (size_t)c * 784 + p];
}

// upsample GEMM out [M0][(i*2+j)*128+d] -> X NHWC [M1,128] (+up_b)
__global__ void upscatter_kernel(const float* __restrict__ tmp,
                                 const float* __restrict__ upb, float* __restrict__ x) {
  const int idx = blockIdx.x * 256 + threadIdx.x;
  if (idx >= M0_ * 512) return;
  const int t = idx >> 9, n = idx & 511;
  const int ij = n >> 7, d = n & 127, i = ij >> 1, j = ij & 1;
  const int b = t / 784, p = t % 784, h0 = p / 28, w0 = p % 28;
  const int row = b * HW_ + (2 * h0 + i) * W_ + (2 * w0 + j);
  x[(size_t)row * C_ + d] = tmp[idx] + upb[d];
}

// NHWC [M1,128] -> NCHW output
__global__ void nhwc2nchw_kernel(const float* __restrict__ x, float* __restrict__ o) {
  const int idx = blockIdx.x * 256 + threadIdx.x;
  if (idx >= B_ * C_ * HW_) return;
  const int p = idx % HW_;
  const int bc = idx / HW_;
  const int c = bc & 127, b = bc >> 7;
  o[idx] = x[((size_t)b * HW_ + p) * C_ + c];
}

// ---------------------------------------------------------------------------
extern "C" void kernel_launch(void* const* d_in, const int* in_sizes, int n_in,
                              void* d_out, int out_size, void* d_ws, size_t ws_size,
                              hipStream_t stream) {
  (void)in_sizes; (void)n_in; (void)out_size;
  const float* F[50];
  for (int i = 0; i < 50; ++i) F[i] = (const float*)d_in[i];
  const int* relIdx = (const int*)d_in[48];
  const float* xin = F[49];

  // per-attention-block parameter indices {blk, grd}
  const int BFC1[2] = {3, 24}, BFC2[2] = {4, 25}, BPRJ[2] = {5, 26}, BQKV[2] = {6, 27};
  const int N1B[2] = {7, 28}, N1G[2] = {8, 29}, N2B[2] = {9, 30}, N2G[2] = {10, 31};
  const int RELB[2] = {11, 32}, WFC1[2] = {12, 33}, WFC2[2] = {13, 34};
  const int WPRJ[2] = {14, 35}, WQKV[2] = {15, 36};

  // ---- workspace layout ----
  size_t off = 0;
  auto alloc = [&](size_t bytes) -> void* {
    off = (off + 255) & ~(size_t)255;
    void* p = (char*)d_ws + off;
    off += bytes;
    return p;
  };
  float*  X    = (float*)alloc((size_t)M1_ * C_ * 4);     // current activation (NHWC)
  float*  T    = (float*)alloc((size_t)M1_ * C_ * 4);     // LN out / attn out
  float*  MIDB = (float*)alloc((size_t)M1_ * MID_ * 4);   // expand / qkv / fc1 hidden
  float*  MID2 = (float*)alloc((size_t)M1_ * MID_ * 4);   // xt / dw out
  float*  POOL = (float*)alloc((size_t)B_ * MID_ * 4);
  float*  SE   = (float*)alloc((size_t)B_ * MID_ * 4);
  float*  B1E  = (float*)alloc(MID_ * 4);
  float*  SC1  = (float*)alloc(MID_ * 4);
  float*  SH1  = (float*)alloc(MID_ * 4);
  float*  SC2  = (float*)alloc(MID_ * 4);
  float*  SH2  = (float*)alloc(MID_ * 4);
  float*  PSC  = (float*)alloc(C_ * 4);
  float*  PSH  = (float*)alloc(C_ * 4);
  float*  BT[2] = {(float*)alloc(HEADS_ * 2401 * 4), (float*)alloc(HEADS_ * 2401 * 4)};
  bf16_t* WUP  = (bf16_t*)alloc((size_t)512 * 256 * 2);
  bf16_t* W1E  = (bf16_t*)alloc((size_t)MID_ * C_ * 2);
  bf16_t* W3B  = (bf16_t*)alloc((size_t)C_ * MID_ * 2);
  bf16_t* QKVB[2] = {(bf16_t*)alloc((size_t)384 * C_ * 2), (bf16_t*)alloc((size_t)384 * C_ * 2)};
  bf16_t* PRJB[2] = {(bf16_t*)alloc((size_t)C_ * C_ * 2), (bf16_t*)alloc((size_t)C_ * C_ * 2)};
  bf16_t* FC1B[2] = {(bf16_t*)alloc((size_t)MID_ * C_ * 2), (bf16_t*)alloc((size_t)MID_ * C_ * 2)};
  bf16_t* FC2B[2] = {(bf16_t*)alloc((size_t)C_ * MID_ * 2), (bf16_t*)alloc((size_t)C_ * MID_ * 2)};
  if (off > ws_size) return;  // workspace too small: nothing safe to do

  auto cdiv = [](int a, int b) { return (a + b - 1) / b; };
  auto cvt = [&](const float* w, bf16_t* o, const float* ksc, int total, int K) {
    cvt_bf16_kernel<<<cdiv(total, 256), 256, 0, stream>>>(w, o, ksc, total, K);
  };
  auto gemm = [&](const float* A, const bf16_t* Wb, float* Out, const float* bias,
                  const float* sc, const float* sh, const float* add,
                  const float* ase, int M, int N, int K, int aseDiv, int g) {
    dim3 grid(M / 128, N / 128);
    if (ase)
      gemm_bf16<true><<<grid, 256, 0, stream>>>(A, Wb, Out, bias, sc, sh, add, ase,
                                                M, N, K, aseDiv, g);
    else
      gemm_bf16<false><<<grid, 256, 0, stream>>>(A, Wb, Out, bias, sc, sh, add, ase,
                                                 M, N, K, aseDiv, g);
  };

  // ---- weight / constant prep ----
  bnvec_kernel<<<1, 256, 0, stream>>>(F[38], F[37], F[39], F[40], PSC, PSH, C_);   // pre_bn
  bnvec_kernel<<<2, 256, 0, stream>>>(F[17], F[16], F[18], F[19], SC1, SH1, MID_); // bn1
  bnvec_kernel<<<2, 256, 0, stream>>>(F[21], F[20], F[22], F[23], SC2, SH2, MID_); // bn2
  b1eff_kernel<<<2, 256, 0, stream>>>(F[43], F[38], F[37], F[39], F[40], B1E);
  upw_pack_kernel<<<cdiv(512 * 256, 256), 256, 0, stream>>>(F[42], WUP);
  cvt(F[43], W1E, PSC, MID_ * C_, C_);   // w1 with pre_bn scale folded in
  cvt(F[44], W3B, nullptr, C_ * MID_, MID_);
  for (int blk = 0; blk < 2; ++blk) {
    cvt(F[WQKV[blk]], QKVB[blk], nullptr, 384 * C_, C_);
    cvt(F[WPRJ[blk]], PRJB[blk], nullptr, C_ * C_, C_);
    cvt(F[WFC1[blk]], FC1B[blk], nullptr, MID_ * C_, C_);
    cvt(F[WFC2[blk]], FC2B[blk], nullptr, C_ * MID_, MID_);
    biastab_kernel<<<cdiv(HEADS_ * 2401, 256), 256, 0, stream>>>(F[RELB[blk]], relIdx,
                                                                 BT[blk]);
  }

  // ---- ConvTranspose2d k=2 s=2 ----
  nchw2nhwc_kernel<<<cdiv(M0_ * CIN_, 256), 256, 0, stream>>>(xin, MID2);
  gemm(MID2, WUP, MIDB, nullptr, nullptr, nullptr, nullptr, nullptr,
       M0_, 512, CIN_, 1, 0);
  upscatter_kernel<<<cdiv(M0_ * 512, 256), 256, 0, stream>>>(MIDB, F[41], X);

  // ---- MBConv ----
  // expand: gelu(bn1(bn_pre(x) @ w1^T))    (pre_bn folded into W1E / B1E)
  gemm(X, W1E, MIDB, B1E, SC1, SH1, nullptr, nullptr, M1_, MID_, C_, 1, 1);
  // depthwise 3x3 + bn2 + gelu
  dwconv_kernel<<<M1_, 256, 0, stream>>>(MIDB, F[45], SC2, SH2, MID2);
  // SE
  pool_kernel<<<dim3(B_, MID_ / 128), 128, 0, stream>>>(MID2, POOL);
  se_kernel<<<B_, 512, 0, stream>>>(POOL, F[46], F[1], F[47], F[2], SE);
  // project (SE scaling folded into A-load) + b3 + residual(shortcut = X)
  gemm(MID2, W3B, X, F[0], nullptr, nullptr, X, SE, M1_, C_, MID_, HW_, 0);

  // ---- two transformer blocks: window (gridPart=0) then grid (gridPart=1) ----
  for (int blk = 0; blk < 2; ++blk) {
    ln_kernel<<<M1_ / 8, 256, 0, stream>>>(X, T, F[N1G[blk]], F[N1B[blk]], M1_);
    gemm(T, QKVB[blk], MIDB, F[BQKV[blk]], nullptr, nullptr, nullptr, nullptr,
         M1_, 384, C_, 1, 0);
    attn_kernel<<<dim3(NWIN_, HEADS_), 128, 0, stream>>>(MIDB, BT[blk], T, blk);
    gemm(T, PRJB[blk], X, F[BPRJ[blk]], nullptr, nullptr, X, nullptr,
         M1_, C_, C_, 1, 0);
    ln_kernel<<<M1_ / 8, 256, 0, stream>>>(X, T, F[N2G[blk]], F[N2B[blk]], M1_);
    gemm(T, FC1B[blk], MIDB, F[BFC1[blk]], nullptr, nullptr, nullptr, nullptr,
         M1_, MID_, C_, 1, 1);
    gemm(MIDB, FC2B[blk], X, F[BFC2[blk]], nullptr, nullptr, X, nullptr,
         M1_, C_, MID_, 1, 0);
  }

  // ---- NHWC -> NCHW output ----
  nhwc2nchw_kernel<<<cdiv(B_ * C_ * HW_, 256), 256, 0, stream>>>(X, (float*)d_out);
}
